// DecoderRNN_38886633898336
// MI455X (gfx1250) — matrix operand order repaired and checked by reading, not compiled, not run on hardware
//
#include <hip/hip_runtime.h>
#include <hip/hip_bf16.h>

// ---------------------------------------------------------------------------
// Problem constants (from the reference): T=100, B=1, IN=36, H=256, E=2048.
// ---------------------------------------------------------------------------
#define TT 100
#define INW 36
#define HH 256
#define EE 2048
#define GG (4 * HH)

typedef __attribute__((ext_vector_type(16))) _Float16 v16h;
typedef __attribute__((ext_vector_type(8)))  float    v8f;

// ---------------------------------------------------------------------------
// Kernel 0: one-shot f32 -> f16 conversion (bandwidth-bound, done once so the
// WMMA kernel moves half the bytes and has zero cvt in its hot loop).
// ---------------------------------------------------------------------------
__global__ __launch_bounds__(256)
void cvt_f16_kernel(const float* __restrict__ src, _Float16* __restrict__ dst,
                    int n)
{
    int i = blockIdx.x * blockDim.x + threadIdx.x;
    int stride = gridDim.x * blockDim.x;
    for (; i < n; i += stride) dst[i] = (_Float16)src[i];
}

// ---------------------------------------------------------------------------
// Kernel A: VOut[h][e] = sum_k att_V[h][k] * outEncoder[e][k]
// (timestep-invariant precompute, 256x2048 output, K=256)
// One wave computes a 16x64 output strip: 1 A-fragment reused over 4 B-frags.
// Operands pre-converted to f16; f32 accumulate via v_wmma_f32_16x16x32_f16.
// ---------------------------------------------------------------------------
__global__ __launch_bounds__(32)
void vout_wmma_kernel(const _Float16* __restrict__ attVh,   // [256][256] f16
                      const _Float16* __restrict__ outEnch, // [2048][256] f16
                      float* __restrict__ VOut)             // [256][2048] f32
{
    const int lane = threadIdx.x;          // 0..31 (wave32)
    const int l15  = lane & 15;
    const int hi   = (lane >> 4) & 1;      // lane half selects K sub-block
    const int m0   = blockIdx.y * 16;      // h tile base
    const int n0   = blockIdx.x * 64;      // e strip base (4 N-tiles)

    v8f acc0 = {}, acc1 = {}, acc2 = {}, acc3 = {};

    for (int k0 = 0; k0 < HH; k0 += 32) {
        // A fragment: 16x32 f16.  ISA layout (16-bit A 16x32):
        //   lanes 0-15:  M=lane,   elems 0..7 -> K=k0+0..7,  8..15 -> K=k0+16..23
        //   lanes 16-31: M=lane-16, elems 0..7 -> K=k0+8..15, 8..15 -> K=k0+24..31
        // Per lane this is two contiguous 8-half (16 B) runs -> b128 loads.
        v16h a;
#pragma unroll
        for (int i = 0; i < 16; ++i) {
            const int k = k0 + (i < 8 ? i : i + 8) + hi * 8;
            a[i] = attVh[(m0 + l15) * HH + k];
        }
        // B fragments: 32x16 f16 each (B[k][e] = outEnc[e][k]).  Same K order as A:
        //   lanes 0-15:  N=lane,   elems 0..7 -> K=k0+0..7,  8..15 -> K=k0+16..23
        //   lanes 16-31: N=lane-16, elems 0..7 -> K=k0+8..15, 8..15 -> K=k0+24..31
        // Per lane: two contiguous 8-half (16 B) runs.
        v16h b0, b1, b2, b3;
#pragma unroll
        for (int i = 0; i < 16; ++i) {
            const int k = k0 + (i < 8 ? i : i + 8) + hi * 8;
            b0[i] = outEnch[(n0 +  0 + l15) * HH + k];
            b1[i] = outEnch[(n0 + 16 + l15) * HH + k];
            b2[i] = outEnch[(n0 + 32 + l15) * HH + k];
            b3[i] = outEnch[(n0 + 48 + l15) * HH + k];
        }
        acc0 = __builtin_amdgcn_wmma_f32_16x16x32_f16(false, a, false, b0,
                                                      (short)0, acc0, false, false);
        acc1 = __builtin_amdgcn_wmma_f32_16x16x32_f16(false, a, false, b1,
                                                      (short)0, acc1, false, false);
        acc2 = __builtin_amdgcn_wmma_f32_16x16x32_f16(false, a, false, b2,
                                                      (short)0, acc2, false, false);
        acc3 = __builtin_amdgcn_wmma_f32_16x16x32_f16(false, a, false, b3,
                                                      (short)0, acc3, false, false);
    }

    // D layout (32-bit C/D 16x16): VGPR r, lanes 0-15 -> M=r, lanes 16-31 -> M=r+8.
#pragma unroll
    for (int r = 0; r < 8; ++r) {
        const int row = m0 + r + hi * 8;
        VOut[row * EE + n0 +  0 + l15] = acc0[r];
        VOut[row * EE + n0 + 16 + l15] = acc1[r];
        VOut[row * EE + n0 + 32 + l15] = acc2[r];
        VOut[row * EE + n0 + 48 + l15] = acc3[r];
    }
}

// ---------------------------------------------------------------------------
// Kernel B: the 100-step sequential scan.  One persistent workgroup of 1024
// threads (32 wave32 waves on a single WGP); all recurrent state in LDS.
// ---------------------------------------------------------------------------
__device__ __forceinline__ float sigmoidf_(float x) {
    return 1.0f / (1.0f + __expf(-x));
}

__global__ __launch_bounds__(1024)
void decoder_seq_kernel(const float* __restrict__ Y,       // [100][1][36]
                        const float* __restrict__ h0,      // [1][1][256]
                        const float* __restrict__ c0,      // [1][1][256]
                        const float* __restrict__ outEnc,  // [2048][256]
                        const float* __restrict__ W_ih,    // [1024][36]
                        const float* __restrict__ W_hh,    // [1024][256]
                        const float* __restrict__ b_ih,    // [1024]
                        const float* __restrict__ b_hh,    // [1024]
                        const float* __restrict__ att_W,   // [256][256]
                        const float* __restrict__ att_b,   // [256][1]
                        const float* __restrict__ att_vec, // [1][256]
                        const float* __restrict__ W1,      // [256][512]
                        const float* __restrict__ b1,      // [256]
                        const float* __restrict__ W2,      // [256][256]
                        const float* __restrict__ b2,      // [256]
                        const float* __restrict__ W3,      // [36][256]
                        const float* __restrict__ b3,      // [36]
                        const float* __restrict__ VOut,    // [256][2048]
                        float* __restrict__ out)           // [3600 + 256]
{
    __shared__ float sh_h[HH], sh_c[HH], sh_hnew[HH], sh_cL[HH];
    __shared__ float sh_WSb[HH], sh_av[HH], sh_ctx[HH], sh_v2[HH];
    __shared__ float sh_gates[GG];
    __shared__ float sh_sc[EE];
    __shared__ float sh_part[1024];
    __shared__ float sh_red[1024];   // reductions, then reused as v1[0:256]

    const int tid = threadIdx.x;

    if (tid < HH) {
        sh_h[tid]  = h0[tid];
        sh_c[tid]  = c0[tid];
        sh_av[tid] = att_vec[tid];
    }
    __syncthreads();

    for (int t = 0; t < TT; ++t) {
        const float* y = Y + t * INW;

        // ---- LSTM gates: gates[j] = y.W_ih[j,:] + h.W_hh[j,:] + b_ih + b_hh
        {
            float g = b_ih[tid] + b_hh[tid];
            const float* wi = W_ih + tid * INW;
#pragma unroll 4
            for (int k = 0; k < INW; ++k) g = fmaf(y[k], wi[k], g);
            const float* wh = W_hh + tid * HH;
#pragma unroll 4
            for (int k = 0; k < HH; ++k) g = fmaf(sh_h[k], wh[k], g);
            sh_gates[tid] = g;
        }
        __syncthreads();

        // ---- cell / hidden update
        if (tid < HH) {
            const float ig = sigmoidf_(sh_gates[tid]);
            const float fg = sigmoidf_(sh_gates[HH + tid]);
            const float gg = tanhf(sh_gates[2 * HH + tid]);
            const float og = sigmoidf_(sh_gates[3 * HH + tid]);
            const float cl = fg * sh_c[tid] + ig * gg;
            sh_cL[tid]   = cl;
            sh_hnew[tid] = og * tanhf(cl);
        }
        __syncthreads();

        // ---- WSb[h] = att_W[h,:].h_new + att_b[h]
        if (tid < HH) {
            float ws = att_b[tid];
            const float* aw = att_W + tid * HH;
#pragma unroll 4
            for (int k = 0; k < HH; ++k) ws = fmaf(aw[k], sh_hnew[k], ws);
            sh_WSb[tid] = ws;
        }
        __syncthreads();

        // ---- scores[e] = sum_h av[h] * tanh(WSb[h] + VOut[h][e]); 2 cols/thread
        {
            const int e0 = tid, e1 = tid + 1024;
            float s0 = 0.0f, s1 = 0.0f;
            for (int h = 0; h < HH; ++h) {
                if ((h & 15) == 0 && h + 16 < HH) {
                    // gfx1250 global_prefetch_b8: pull rows a few iters ahead
                    __builtin_prefetch(&VOut[(h + 16) * EE + e0], 0, 3);
                }
                const float* vr = VOut + h * EE;
                const float w  = sh_WSb[h];
                const float vv = sh_av[h];
                s0 = fmaf(vv, tanhf(w + vr[e0]), s0);
                s1 = fmaf(vv, tanhf(w + vr[e1]), s1);
            }
            sh_sc[e0] = s0;
            sh_sc[e1] = s1;
        }
        __syncthreads();

        // ---- softmax over 2048 (LDS tree reductions)
        {
            sh_red[tid] = fmaxf(sh_sc[tid], sh_sc[tid + 1024]);
            __syncthreads();
            for (int off = 512; off > 0; off >>= 1) {
                if (tid < off) sh_red[tid] = fmaxf(sh_red[tid], sh_red[tid + off]);
                __syncthreads();
            }
            const float mx = sh_red[0];
            __syncthreads();
            const float e0v = __expf(sh_sc[tid] - mx);
            const float e1v = __expf(sh_sc[tid + 1024] - mx);
            sh_sc[tid] = e0v;
            sh_sc[tid + 1024] = e1v;
            sh_red[tid] = e0v + e1v;
            __syncthreads();
            for (int off = 512; off > 0; off >>= 1) {
                if (tid < off) sh_red[tid] += sh_red[tid + off];
                __syncthreads();
            }
            const float inv = 1.0f / sh_red[0];
            __syncthreads();
            sh_sc[tid]        *= inv;
            sh_sc[tid + 1024] *= inv;
        }
        __syncthreads();

        // ---- ctx[h] = sum_e alpha[e] * outEnc[e][h]; 4 partial threads per h
        {
            const int hh   = tid & (HH - 1);
            const int part = tid >> 8;           // 0..3
            float s = 0.0f;
            const int eb = part * 512;
            for (int e = eb; e < eb + 512; ++e)
                s = fmaf(sh_sc[e], outEnc[e * HH + hh], s);
            sh_part[part * HH + hh] = s;
        }
        __syncthreads();
        if (tid < HH) {
            sh_ctx[tid] = sh_part[tid] + sh_part[HH + tid] +
                          sh_part[2 * HH + tid] + sh_part[3 * HH + tid];
        }
        __syncthreads();

        // ---- MLP: v = [cLSTM, ctx];  v1 = relu(v.W1^T + b1)  (v1 in sh_red)
        if (tid < HH) {
            float s = b1[tid];
            const float* w = W1 + tid * (2 * HH);
#pragma unroll 4
            for (int k = 0; k < HH; ++k) s = fmaf(w[k], sh_cL[k], s);
#pragma unroll 4
            for (int k = 0; k < HH; ++k) s = fmaf(w[HH + k], sh_ctx[k], s);
            sh_red[tid] = fmaxf(s, 0.0f);
        }
        __syncthreads();
        if (tid < HH) {
            float s = b2[tid];
            const float* w = W2 + tid * HH;
#pragma unroll 4
            for (int k = 0; k < HH; ++k) s = fmaf(w[k], sh_red[k], s);
            sh_v2[tid] = fmaxf(s, 0.0f);
        }
        __syncthreads();
        if (tid < INW) {
            float s = b3[tid];
            const float* w = W3 + tid * HH;
#pragma unroll 4
            for (int k = 0; k < HH; ++k) s = fmaf(w[k], sh_v2[k], s);
            out[t * INW + tid] = s;
        }

        // ---- carry update: h <- h_new, c <- ctx (reference feeds ctx as cell)
        if (tid < HH) {
            sh_h[tid] = sh_hnew[tid];
            sh_c[tid] = sh_ctx[tid];
            if (t == TT - 1) out[TT * INW + tid] = sh_ctx[tid];  // c_final
        }
        __syncthreads();
    }
}

// ---------------------------------------------------------------------------
// Host entry point
// ---------------------------------------------------------------------------
extern "C" void kernel_launch(void* const* d_in, const int* in_sizes, int n_in,
                              void* d_out, int out_size, void* d_ws, size_t ws_size,
                              hipStream_t stream) {
    (void)in_sizes; (void)n_in; (void)out_size; (void)ws_size;

    const float* Y       = (const float*)d_in[0];
    const float* h0      = (const float*)d_in[1];
    const float* c0      = (const float*)d_in[2];
    const float* outEnc  = (const float*)d_in[3];
    const float* W_ih    = (const float*)d_in[4];
    const float* W_hh    = (const float*)d_in[5];
    const float* b_ih    = (const float*)d_in[6];
    const float* b_hh    = (const float*)d_in[7];
    const float* att_W   = (const float*)d_in[8];
    const float* att_V   = (const float*)d_in[9];
    const float* att_b   = (const float*)d_in[10];
    const float* att_vec = (const float*)d_in[11];
    const float* W1      = (const float*)d_in[12];
    const float* b1      = (const float*)d_in[13];
    const float* W2      = (const float*)d_in[14];
    const float* b2      = (const float*)d_in[15];
    const float* W3      = (const float*)d_in[16];
    const float* b3      = (const float*)d_in[17];

    // Workspace layout:
    //   [0, 2MB)          VOut      f32 [256][2048]
    //   [2MB, 2MB+128KB)  attV_h    f16 [256][256]
    //   [2MB+128KB, +1MB) outEnc_h  f16 [2048][256]
    char* ws = (char*)d_ws;
    float*     VOut     = (float*)ws;
    _Float16*  attVh    = (_Float16*)(ws + (size_t)HH * EE * sizeof(float));
    _Float16*  outEnch  = (_Float16*)(ws + (size_t)HH * EE * sizeof(float)
                                         + (size_t)HH * HH * sizeof(_Float16));
    float* out = (float*)d_out;        // 3600 outputs + 256 c_final

    // One-shot f16 conversion of the GEMM operands (bandwidth trivial).
    cvt_f16_kernel<<<64, 256, 0, stream>>>(att_V, attVh, HH * HH);
    cvt_f16_kernel<<<256, 256, 0, stream>>>(outEnc, outEnch, EE * HH);

    // Wide WMMA precompute of the timestep-invariant attention table.
    dim3 gridA(EE / 64, HH / 16);      // (32, 16) waves
    vout_wmma_kernel<<<gridA, 32, 0, stream>>>(attVh, outEnch, VOut);

    // Persistent single-WGP sequential scan (depends on VOut; same stream).
    decoder_seq_kernel<<<1, 1024, 0, stream>>>(Y, h0, c0, outEnc,
                                               W_ih, W_hh, b_ih, b_hh,
                                               att_W, att_b, att_vec,
                                               W1, b1, W2, b2, W3, b3,
                                               VOut, out);
}
